// TextLSTM_60524679135340
// MI455X (gfx1250) — matrix operand order repaired and checked by reading, not compiled, not run on hardware
//
#include <hip/hip_runtime.h>

// ---------------------------------------------------------------------------
// TextLSTM on MI455X (gfx1250): 2-layer LSTM reference, but layer 2 is dead
// code (h1/c1 never affect the output), so we compute only:
//   1) Xbf  = bf16(emb[X])                              (gather+convert)
//   2) XW   = Xbf @ Wi + (bi+bh)        [16384x2048]    (bf16 WMMA, f32 out)
//   3) scan: pre = XW[t] + h @ Wh ; gates ; c,h update  (persistent kernel)
//   4) out  = h @ Wout^T + bout         [64x50257]      (bf16 WMMA, f32 out)
// GEMMs: v_wmma_f32_16x16x32_bf16. Weight staging: global_load_async_to_lds.
// ---------------------------------------------------------------------------

#define BATCH   64
#define TSTEPS  256
#define EMB     512
#define HID     512
#define GATES   2048        // 4*HID, gate order [i, f, g, o]
#define NCLASS  50257
#define NPAD    50304       // 393 * 128 (pad N for the logits GEMM)

typedef __bf16 bf16_t;
typedef __attribute__((ext_vector_type(16))) __bf16 v16bf;
typedef __attribute__((ext_vector_type(8)))  __bf16 v8bf;
typedef __attribute__((ext_vector_type(8)))  float  v8f;
typedef __attribute__((ext_vector_type(4)))  int    v4i;

// AS-qualified pointer types for the async-to-LDS builtin
// (signature per hipcc diagnostic: (v4i AS1*, v4i AS3*, imm, imm))
typedef __attribute__((address_space(1))) v4i*  gv4p;
typedef __attribute__((address_space(3))) v4i*  lv4p;
typedef __attribute__((address_space(1))) void* gvoidp;
typedef __attribute__((address_space(3))) void* lvoidp;

__device__ __forceinline__ v8f wmma_bf16(v16bf a, v16bf b, v8f c) {
  // (neg_a, A, neg_b, B, c_mod, C, reuse_a, reuse_b)
  return __builtin_amdgcn_wmma_f32_16x16x32_bf16(false, a, false, b,
                                                 (short)0, c, false, false);
}

// A fragment, 16x32 bf16 tile from row-major matrix:
// lane L holds row M=L%16; K values: (L/16)*16+j (matches B fragment order)
// => two contiguous 8-half (16B) chunks per lane.
__device__ __forceinline__ v16bf load_a16x32(const bf16_t* __restrict__ tile,
                                             int ld, int lane) {
  const bf16_t* row = tile + (size_t)(lane & 15) * ld + ((lane >> 4) << 4);
  v8bf lo = *(const v8bf*)(row);
  v8bf hi = *(const v8bf*)(row + 8);
  v16bf a;
#pragma unroll
  for (int j = 0; j < 8; ++j) { a[j] = lo[j]; a[j + 8] = hi[j]; }
  return a;
}

// B fragment map (32x16 bf16): lane L holds col N=L%16, K = (L/16)*16 + j.
// Weights are pre-swizzled so each lane's 16 halves are one aligned 32B load.

// Async-stage `n128` 16-byte chunks from global to LDS (ASYNCcnt path on
// gfx1250; falls back to a synchronous copy if the builtin is unavailable).
__device__ __forceinline__ void stage_to_lds(const bf16_t* __restrict__ src,
                                             bf16_t* __restrict__ lds,
                                             int n128, int tid, int nthreads) {
#if __has_builtin(__builtin_amdgcn_global_load_async_to_lds_b128)
  for (int i = tid; i < n128; i += nthreads) {
    __builtin_amdgcn_global_load_async_to_lds_b128(
        (gv4p)(gvoidp)(src + (size_t)i * 8),
        (lv4p)(lvoidp)(lds + (size_t)i * 8),
        0, 0);
  }
#if __has_builtin(__builtin_amdgcn_s_wait_asynccnt)
  __builtin_amdgcn_s_wait_asynccnt(0);
#else
  asm volatile("s_wait_asynccnt 0" ::: "memory");
#endif
#else
  const uint4* s = (const uint4*)src;
  uint4* d = (uint4*)lds;
  for (int i = tid; i < n128; i += nthreads) d[i] = s[i];
#endif
}

// ------------------------------ prep kernels -------------------------------

__global__ void k_zero32(uint32_t* __restrict__ p, int n) {
  int i = blockIdx.x * blockDim.x + threadIdx.x;
  if (i < n) p[i] = 0u;
}

// Xbf[(t*64+b)][e] = bf16(emb[X[b][t]][e])
__global__ void k_prep_x(const int* __restrict__ X, const float* __restrict__ emb,
                         bf16_t* __restrict__ Xbf) {
  size_t idx = (size_t)blockIdx.x * blockDim.x + threadIdx.x;
  if (idx >= (size_t)BATCH * TSTEPS * EMB) return;
  int e = (int)(idx & (EMB - 1));
  int r = (int)(idx >> 9);
  int t = r >> 6, b = r & 63;
  int tok = X[b * TSTEPS + t];
  Xbf[idx] = (bf16_t)emb[(size_t)tok * EMB + e];
}

// WiFrag[nt][kk][lane][j]  (nt:128, kk:16, lane:32, j:16), B-fragment order.
__global__ void k_prep_wi(const float* __restrict__ Wi, bf16_t* __restrict__ WiFrag) {
  int idx = blockIdx.x * blockDim.x + threadIdx.x;
  if (idx >= 128 * 16 * 32 * 16) return;
  int j = idx & 15, lane = (idx >> 4) & 31, kk = (idx >> 9) & 15, nt = idx >> 13;
  int k = kk * 32 + ((lane >> 4) << 4) + j;
  int g = nt * 16 + (lane & 15);
  WiFrag[idx] = (bf16_t)Wi[(size_t)k * GATES + g];
}

// WhFrag[blk][q][kk][lane][j]  (blk:32, q:4 gates, kk:16, lane:32, j:16).
// Block blk owns h columns [16*blk, 16*blk+16); its B matrix packs the four
// gate strips q*512 + blk*16 + n.
__global__ void k_prep_wh(const float* __restrict__ Wh, bf16_t* __restrict__ WhFrag) {
  int idx = blockIdx.x * blockDim.x + threadIdx.x;
  if (idx >= 32 * 4 * 16 * 32 * 16) return;
  int j = idx & 15, lane = (idx >> 4) & 31, kk = (idx >> 9) & 15;
  int q = (idx >> 13) & 3, blk = idx >> 15;
  int k = kk * 32 + ((lane >> 4) << 4) + j;
  int col = q * HID + blk * 16 + (lane & 15);
  WhFrag[idx] = (bf16_t)Wh[(size_t)k * GATES + col];
}

// WoutBf: row-major [NPAD][HID] bf16, zero-padded rows >= NCLASS.
__global__ void k_prep_wout(const float* __restrict__ Wout, bf16_t* __restrict__ WoutBf) {
  size_t idx = (size_t)blockIdx.x * blockDim.x + threadIdx.x;
  if (idx >= (size_t)NPAD * HID) return;
  int k = (int)(idx & (HID - 1));
  int n = (int)(idx >> 9);
  WoutBf[idx] = (n < NCLASS) ? (bf16_t)Wout[(size_t)n * HID + k] : (bf16_t)0.f;
}

// ------------------------- input-projection GEMM ---------------------------
// XW[r][g] = Xbf[r] . Wi[:,g] + bi[g] + bh[g];  r = t*64+b, 16384 x 2048.
// Block = 128x64 tile, 8 waves; wave = one 16-row M tile x four N tiles.
// The block's 64 KB Wi-fragment slice (4 N-tiles, all K) is async-staged into
// LDS once and shared by all 8 waves (B operands come from ds_load_b128).
__global__ __launch_bounds__(256) void k_xproj(
    const bf16_t* __restrict__ Xbf, const bf16_t* __restrict__ WiFrag,
    const float* __restrict__ bi, const float* __restrict__ bh,
    float* __restrict__ XW) {
  __shared__ __align__(16) bf16_t wlds[4 * 16 * 32 * 16];  // 64 KB
  const int lane = threadIdx.x & 31;
  const int mt   = threadIdx.x >> 5;          // 0..7
  const int rb   = (blockIdx.x >> 5) * 128;
  const int cb   = (blockIdx.x & 31) * 64;

  // NT slabs are contiguous in WiFrag (nt-major): stage NT in [cb/16, cb/16+4)
  stage_to_lds(WiFrag + (size_t)(cb >> 4) * (16 * 32 * 16), wlds, 4096,
               threadIdx.x, 256);
  __syncthreads();

  v8f acc[4];
#pragma unroll
  for (int q = 0; q < 4; ++q)
#pragma unroll
    for (int v = 0; v < 8; ++v) acc[q][v] = 0.f;

  for (int kk = 0; kk < 16; ++kk) {
    v16bf a = load_a16x32(Xbf + (size_t)(rb + mt * 16) * EMB + kk * 32, EMB, lane);
#pragma unroll
    for (int nt = 0; nt < 4; ++nt) {
      v16bf b = *(const v16bf*)(wlds + (((nt * 16 + kk) * 32 + lane) << 4));
      acc[nt] = wmma_bf16(a, b, acc[nt]);
    }
  }

  const int jloc = lane & 15, rsub = (lane >> 4) << 3;
#pragma unroll
  for (int nt = 0; nt < 4; ++nt) {
    const int g = cb + nt * 16 + jloc;
    const float bias = bi[g] + bh[g];
#pragma unroll
    for (int v = 0; v < 8; ++v) {
      const int row = rb + mt * 16 + rsub + v;
      XW[(size_t)row * GATES + g] = acc[nt][v] + bias;
    }
  }
}

// ----------------------------- recurrent scan ------------------------------

__device__ __forceinline__ void grid_sync(int* sync, int target) {
  __builtin_amdgcn_fence(__ATOMIC_RELEASE, "agent");   // flush h stores to DEV scope
  __syncthreads();
  if (threadIdx.x == 0) {
    __hip_atomic_fetch_add(sync, 1, __ATOMIC_RELAXED, __HIP_MEMORY_SCOPE_AGENT);
    while (__hip_atomic_load(sync, __ATOMIC_RELAXED, __HIP_MEMORY_SCOPE_AGENT) < target)
      __builtin_amdgcn_s_sleep(2);
  }
  __syncthreads();
  __builtin_amdgcn_fence(__ATOMIC_ACQUIRE, "agent");   // invalidate stale h lines
}

// 32 persistent blocks x 128 threads (4 wave32). Block blk owns h cols
// [16blk,16blk+16); wave mt owns rows [16mt,16mt+16). Each wave accumulates
// all four gate tiles for its (rows, cols) patch, so the cell update is
// lane-local; c lives in 8 VGPRs/lane. Wh slice pinned in LDS (64 KB),
// staged once with async loads and reused for all 256 timesteps.
__global__ __launch_bounds__(128) void k_lstm_scan(
    const bf16_t* __restrict__ WhFrag, const float* __restrict__ XW,
    bf16_t* __restrict__ hbuf0, bf16_t* __restrict__ hbuf1,
    int* __restrict__ sync) {
  __shared__ __align__(16) bf16_t whlds[4 * 16 * 32 * 16];  // 64 KB
  const int blk  = blockIdx.x;
  const int lane = threadIdx.x & 31;
  const int mt   = threadIdx.x >> 5;

  stage_to_lds(WhFrag + (size_t)blk * (4 * 16 * 32 * 16), whlds, 4096,
               threadIdx.x, 128);
  __syncthreads();

  float c[8];
#pragma unroll
  for (int v = 0; v < 8; ++v) c[v] = 0.f;

  const int jloc = lane & 15, rsub = (lane >> 4) << 3;

  for (int t = 0; t < TSTEPS; ++t) {
    const bf16_t* hr = (t & 1) ? hbuf1 : hbuf0;   // read h(t-1)
    bf16_t*       hw = (t & 1) ? hbuf0 : hbuf1;   // write h(t); t=255 -> hbuf0

    v8f acc[4];
#pragma unroll
    for (int q = 0; q < 4; ++q)
#pragma unroll
      for (int v = 0; v < 8; ++v) acc[q][v] = 0.f;

#pragma unroll 4
    for (int kk = 0; kk < 16; ++kk) {
      v16bf a = load_a16x32(hr + (size_t)(mt * 16) * HID + kk * 32, HID, lane);
#pragma unroll
      for (int q = 0; q < 4; ++q) {
        v16bf b = *(const v16bf*)(whlds + (((q * 16 + kk) * 32 + lane) << 4));
        acc[q] = wmma_bf16(a, b, acc[q]);
      }
    }

    const size_t xwbase =
        ((size_t)t * BATCH + mt * 16 + rsub) * GATES + blk * 16 + jloc;
#pragma unroll
    for (int v = 0; v < 8; ++v) {
      const size_t xr = xwbase + (size_t)v * GATES;
      float pi = acc[0][v] + XW[xr + 0 * HID];
      float pf = acc[1][v] + XW[xr + 1 * HID];
      float pg = acc[2][v] + XW[xr + 2 * HID];
      float po = acc[3][v] + XW[xr + 3 * HID];
      float iv = 1.f / (1.f + __expf(-pi));
      float fv = 1.f / (1.f + __expf(-pf));
      float gv = tanhf(pg);
      float ov = 1.f / (1.f + __expf(-po));
      c[v] = fv * c[v] + iv * gv;
      float hv = ov * tanhf(c[v]);
      hw[(size_t)(mt * 16 + rsub + v) * HID + blk * 16 + jloc] = (bf16_t)hv;
    }
    grid_sync(sync, 32 * (t + 1));
  }
}

// ------------------------------- logits GEMM -------------------------------
// out[r][n] = h[r] . Wout[n] + bout[n].  Bandwidth-bound on Wout (51 MB bf16).
// Block = 64x128 tile, 8 waves; wave = one 16-col N tile x four M tiles.
__global__ __launch_bounds__(256) void k_logits(
    const bf16_t* __restrict__ hfin, const bf16_t* __restrict__ WoutBf,
    const float* __restrict__ bout, float* __restrict__ out) {
  const int lane = threadIdx.x & 31;
  const int w    = threadIdx.x >> 5;
  const int n0   = blockIdx.x * 128 + w * 16;

  v8f acc[4];
#pragma unroll
  for (int q = 0; q < 4; ++q)
#pragma unroll
    for (int v = 0; v < 8; ++v) acc[q][v] = 0.f;

  for (int kk = 0; kk < 16; ++kk) {
    // B fragment: lane holds col n0+(lane&15); K = kk*32 + (lane>>4)*16 + j
    // -> 16 contiguous halves of the row-major Wout row (one 32B load).
    const bf16_t* bp =
        WoutBf + (size_t)(n0 + (lane & 15)) * HID + kk * 32 + ((lane >> 4) << 4);
    v16bf b = *(const v16bf*)bp;
    __builtin_prefetch(bp + 32, 0, 0);  // stream next K chunk (global_prefetch_b8)
#pragma unroll
    for (int mt = 0; mt < 4; ++mt) {
      v16bf a = load_a16x32(hfin + (size_t)(mt * 16) * HID + kk * 32, HID, lane);
      acc[mt] = wmma_bf16(a, b, acc[mt]);
    }
  }

  const int n = n0 + (lane & 15);
  if (n < NCLASS) {
    const float bb = bout[n];
    const int rsub = (lane >> 4) << 3;
#pragma unroll
    for (int mt = 0; mt < 4; ++mt)
#pragma unroll
      for (int v = 0; v < 8; ++v)
        out[(size_t)(mt * 16 + rsub + v) * NCLASS + n] = acc[mt][v] + bb;
  }
}

// ------------------------------- launcher ----------------------------------

extern "C" void kernel_launch(void* const* d_in, const int* in_sizes, int n_in,
                              void* d_out, int out_size, void* d_ws, size_t ws_size,
                              hipStream_t stream) {
  const int*   X    = (const int*)d_in[0];
  const float* emb  = (const float*)d_in[1];
  const float* Wi   = (const float*)d_in[2];
  const float* Wh   = (const float*)d_in[3];
  const float* bi   = (const float*)d_in[4];
  const float* bh   = (const float*)d_in[5];
  // d_in[6..9]: layer-2 weights — dead code, never used by the output.
  const float* Wout = (const float*)d_in[10];
  const float* bout = (const float*)d_in[11];
  float* out = (float*)d_out;

  char* ws = (char*)d_ws;
  size_t o = 0;
  auto alloc = [&](size_t bytes) {
    size_t r = o;
    o += (bytes + 255) & ~(size_t)255;
    return r;
  };
  bf16_t* WiFrag = (bf16_t*)(ws + alloc((size_t)EMB * GATES * 2));        //  2 MB
  bf16_t* WhFrag = (bf16_t*)(ws + alloc((size_t)HID * GATES * 2));        //  2 MB
  bf16_t* WoutBf = (bf16_t*)(ws + alloc((size_t)NPAD * HID * 2));         // 51 MB
  bf16_t* Xbf    = (bf16_t*)(ws + alloc((size_t)BATCH * TSTEPS * EMB * 2));// 16 MB
  float*  XW     = (float*)(ws + alloc((size_t)BATCH * TSTEPS * GATES * 4));//128 MB
  bf16_t* hbuf0  = (bf16_t*)(ws + alloc((size_t)BATCH * HID * 2));        // 64 KB
  bf16_t* hbuf1  = (bf16_t*)(ws + alloc((size_t)BATCH * HID * 2));        // 64 KB
  int*    syncp  = (int*)(ws + alloc(256));

  // zero h buffers + sync counter (hbuf0/hbuf1/sync are contiguous)
  {
    int words = (int)(((size_t)BATCH * HID * 2 * 2 + 256) / 4);
    k_zero32<<<(words + 255) / 256, 256, 0, stream>>>((uint32_t*)hbuf0, words);
  }
  k_prep_x<<<(BATCH * TSTEPS * EMB + 255) / 256, 256, 0, stream>>>(X, emb, Xbf);
  k_prep_wi<<<(128 * 16 * 32 * 16 + 255) / 256, 256, 0, stream>>>(Wi, WiFrag);
  k_prep_wh<<<(32 * 4 * 16 * 32 * 16 + 255) / 256, 256, 0, stream>>>(Wh, WhFrag);
  k_prep_wout<<<((size_t)NPAD * HID + 255) / 256, 256, 0, stream>>>(Wout, WoutBf);

  // input projection: 16384x2048 over K=512; 128x64 tiles
  k_xproj<<<dim3(128 * 32), dim3(256), 0, stream>>>(Xbf, WiFrag, bi, bh, XW);

  // persistent recurrent scan: 32 blocks x 128 threads, 256 steps
  k_lstm_scan<<<dim3(32), dim3(128), 0, stream>>>(WhFrag, XW, hbuf0, hbuf1, syncp);

  // logits: 64 x 50257 (padded 50304), final h is in hbuf0
  k_logits<<<dim3(NPAD / 128), dim3(256), 0, stream>>>(hbuf0, WoutBf, bout, out);
}
